// Network_18769007083781
// MI455X (gfx1250) — matrix operand order repaired and checked by reading, not compiled, not run on hardware
//
#include <hip/hip_runtime.h>
#include <hip/hip_bf16.h>

typedef _Float16 f16;
typedef _Float16 v16h __attribute__((ext_vector_type(16)));
typedef float    v8f  __attribute__((ext_vector_type(8)));
typedef unsigned u32x4 __attribute__((ext_vector_type(4)));
typedef int      i32x4 __attribute__((ext_vector_type(4)));
typedef int      i32x8 __attribute__((ext_vector_type(8)));

#define N_RAYS 2048
#define SAMP   84
#define NPTS   (N_RAYS*SAMP)       /* 172032 */
#define TILE_M 128
#define NTILES (NPTS/TILE_M)       /* 1344 exact */
#define DISTC  200.0f

/* packed weights: pair-packed f16, row stride (2*Npad+8) f16 = (Npad+4) dwords
   so the two lane-halves of a B fragment hit disjoint LDS banks (8*S%64==32) */
#define PW_SZ(Kpad,Npad)  ((Kpad/2)*(2*(Npad)+8))
#define OFF_W_IN    0
#define SZ_W_IN     PW_SZ(64,256)          /* 16640  */
#define OFF_W_H     (OFF_W_IN + SZ_W_IN)
#define SZ_W_H1     PW_SZ(256,256)         /* 66560  */
#define OFF_W_SKIP  (OFF_W_H + 6*SZ_W_H1)
#define SZ_W_SKIP   PW_SZ(320,256)         /* 83200  */
#define OFF_W_SS    (OFF_W_SKIP + SZ_W_SKIP)
#define SZ_W_SS     PW_SZ(256,64)          /* 17408  */
#define OFF_W_FEAT  (OFF_W_SS + SZ_W_SS)
#define SZ_W_FEAT   PW_SZ(256,256)         /* 66560  */
#define OFF_W_RGB1  (OFF_W_FEAT + SZ_W_FEAT)
#define SZ_W_RGB1   PW_SZ(288,128)         /* 38016  */
#define OFF_W_RGB2  (OFF_W_RGB1 + SZ_W_RGB1)
#define SZ_W_RGB2   PW_SZ(128,16)          /* 2560   */
#define W_TOTAL     (OFF_W_RGB2 + SZ_W_RGB2)

/* LDS layout (f16 elems): padded activation rows kill bank conflicts */
#define X0_LD   72
#define H_LD    264
#define X0_ELE  (128*X0_LD)                /* 9216  */
#define H_ELE   (128*H_LD)                 /* 33792 */
#define WLDS_OFF_BYTES ((X0_ELE + 2*H_ELE)*2)   /* 153600 */
#define SMEM_BYTES (WLDS_OFF_BYTES + 2*SZ_W_SKIP) /* +166400 = 320000 */

/* =================== weight conversion: f32 -> pair-packed f16 ============ */
__global__ void pack_w(const float* __restrict__ src, f16* __restrict__ dst,
                       int Ks, int Ns, int Npad, int noff)
{
    int i = blockIdx.x * blockDim.x + threadIdx.x;
    if (i >= Ks * Ns) return;
    int k = i / Ns;
    int n = i % Ns + noff;
    dst[(size_t)(k >> 1) * (2*Npad + 8) + 2*n + (k & 1)] = (f16)src[i];
}

/* =================== per-ray sampling + posenc ===================== */
__global__ __launch_bounds__(256) void sample_kernel(
    const float* __restrict__ rays, const float* __restrict__ inter,
    const float* __restrict__ noise,
    f16* __restrict__ act0, f16* __restrict__ denc, float* __restrict__ zout)
{
    __shared__ float zf[256];
    __shared__ float za[128];
    const int r = blockIdx.x;
    const int t = threadIdx.x;
    const float* rb = rays + (size_t)r * 6;
    const float dx = rb[3], dy = rb[4], dz = rb[5];
    const float scale = sqrtf(dx*dx + dy*dy + dz*dz);

    if (t < 240) {
        int b = t / 24, j = t % 24;
        float nb = inter[((size_t)r*10 + b)*4 + 0];
        float fb = inter[((size_t)r*10 + b)*4 + 1];
        zf[t] = nb + (fb - nb) * (j * (1.0f/23.0f));
    } else zf[t] = 3.0e38f;
    __syncthreads();

    for (int k = 2; k <= 256; k <<= 1)
        for (int j = k >> 1; j > 0; j >>= 1) {
            int ixj = t ^ j;
            if (ixj > t) {
                float a = zf[t], b = zf[ixj];
                bool up = ((t & k) == 0);
                if ((a > b) == up) { zf[t] = b; zf[ixj] = a; }
            }
            __syncthreads();
        }

    float zval = 0.0f;
    if (t < 64) {
        float u   = t * (1.0f/63.0f);
        int inds  = (int)floorf(u * 239.0f) + 1;
        int below = min(inds - 1, 239);
        int above = min(inds, 239);
        float cb = below * (1.0f/239.0f);
        float ca = above * (1.0f/239.0f);
        float den = ca - cb; if (den < 1e-5f) den = 1.0f;
        float tt = (u - cb) / den;
        zval = zf[below] + tt * (zf[above] - zf[below]);
    } else if (t < 84) {
        int q = t - 64;
        if (q < 10) zval = inter[((size_t)r*10 + q)*4 + 0] - 1e-5f;
        else        zval = inter[((size_t)r*10 + (q-10))*4 + 1] + 1e-5f;
    }
    if (t < 84 && zval < 0.0f) zval = DISTC + 20.0f * noise[(size_t)r*SAMP + t];
    __syncthreads();
    if (t < 128) za[t] = (t < 84) ? zval : 3.0e38f;
    __syncthreads();

    for (int k = 2; k <= 128; k <<= 1)
        for (int j = k >> 1; j > 0; j >>= 1) {
            if (t < 128) {
                int ixj = t ^ j;
                if (ixj > t) {
                    float a = za[t], b = za[ixj];
                    bool up = ((t & k) == 0);
                    if ((a > b) == up) { za[t] = b; za[ixj] = a; }
                }
            }
            __syncthreads();
        }

    if (t < 84) {
        float z = za[t];
        zout[(size_t)r*SAMP + t] = z;
        size_t p = (size_t)r*SAMP + t;
        f16* o = act0 + p * 64;
        float xs = (rb[0] + dx * z / scale) * (1.0f/DISTC);
        float ys = (rb[1] + dy * z / scale) * (1.0f/DISTC);
        float zs = (rb[2] + dz * z / scale) * (1.0f/DISTC);
        o[0] = (f16)xs; o[1] = (f16)ys; o[2] = (f16)zs;
        float f = 1.0f;
        for (int l = 0; l < 10; l++) {
            o[3 + 6*l + 0] = (f16)sinf(xs * f);
            o[3 + 6*l + 1] = (f16)sinf(ys * f);
            o[3 + 6*l + 2] = (f16)sinf(zs * f);
            o[3 + 6*l + 3] = (f16)cosf(xs * f);
            o[3 + 6*l + 4] = (f16)cosf(ys * f);
            o[3 + 6*l + 5] = (f16)cosf(zs * f);
            f *= 2.0f;
        }
        o[63] = (f16)0.0f;
    }
    if (t >= 96 && t < 128) {
        int c = t - 96;
        float v = 0.0f;
        float ix = dx / scale, iy = dy / scale, iz = dz / scale;
        if (c < 3) v = (c == 0) ? ix : ((c == 1) ? iy : iz);
        else if (c < 27) {
            int q = c - 3, l = q / 6, w = q % 6;
            float comp = ((w % 3) == 0) ? ix : (((w % 3) == 1) ? iy : iz);
            float f = (float)(1 << l);
            v = (w < 3) ? sinf(comp * f) : cosf(comp * f);
        }
        denc[(size_t)r*32 + c] = (f16)v;
    }
}

/* =================== TDM staging: global -> LDS ===================== */
__device__ __forceinline__ void stage_weights(const f16* __restrict__ Wg,
                                              f16* __restrict__ Wl,
                                              unsigned lds_off_bytes,
                                              unsigned nbytes, int tid)
{
#if __has_builtin(__builtin_amdgcn_tensor_load_to_lds)
    if (tid < 32) {                     /* wave 0 issues; TDM ignores EXEC */
        unsigned nunits = nbytes >> 3;  /* data_size = 8B units */
        unsigned long long ga = (unsigned long long)(uintptr_t)Wg;
        u32x4 g0;
        g0[0] = 1u;                                            /* count=1 */
        g0[1] = lds_off_bytes;                                 /* lds_addr */
        g0[2] = (unsigned)ga;                                  /* gaddr lo */
        g0[3] = (unsigned)((ga >> 32) & 0x01FFFFFFull) | (2u << 30); /* type=2 */
        i32x8 g1;
        g1[0] = 0x00030000;                                    /* data_size=8B */
        g1[1] = (int)((nunits & 0xFFFFu) << 16);               /* tensor_dim0 lo */
        g1[2] = (int)(((nunits >> 16) & 0xFFFFu) | (1u << 16));/* dim0 hi | dim1=1 */
        g1[3] = (int)((nunits & 0xFFFFu) << 16);               /* tile_dim0 */
        g1[4] = 1;                                             /* tile_dim1=1 */
        g1[5] = (int)nunits;                                   /* dim0_stride lo */
        g1[6] = 0;
        g1[7] = 0;
        i32x4 z4 = {0,0,0,0};
# if __has_include(<hip/amd_detail/amd_gfx1250_TDM.h>)
        i32x8 z8 = {0,0,0,0,0,0,0,0};
        __builtin_amdgcn_tensor_load_to_lds(g0, g1, z4, z4, z8, 0);
# else
        __builtin_amdgcn_tensor_load_to_lds(g0, g1, z4, z4, 0);
# endif
        __builtin_amdgcn_s_wait_tensorcnt(0);
    }
    (void)Wl;
#else
    const uint4* s = (const uint4*)Wg;
    uint4* d = (uint4*)Wl;
    for (unsigned i = tid; i < nbytes / 16; i += 256) d[i] = s[i];
#endif
    __syncthreads();
}

/* =================== WMMA fragment loaders ===================== */
__device__ __forceinline__ v16h frag_a(const f16* act, int ld, int row, int k0, int hf)
{
    union { v16h v; unsigned u[8]; } A;
    const unsigned* p = (const unsigned*)(act + (size_t)row * ld);
#pragma unroll
    for (int i = 0; i < 8; i++) {
        int pk = 2*i + 16*hf;
        A.u[i] = p[(k0 + pk) >> 1];
    }
    return A.v;
}
/* B fragment from pair-packed LDS weights, dword row stride S = Npad+4 */
__device__ __forceinline__ v16h frag_bs(const unsigned* W32, int S, int k0, int col, int hf)
{
    union { v16h v; unsigned u[8]; } B;
    int rbase = (k0 + 16*hf) >> 1;
#pragma unroll
    for (int i = 0; i < 8; i++)
        B.u[i] = W32[(size_t)(rbase + i) * S + col];
    return B.v;
}

/* dense layer with A fragments cached in registers across all column tiles */
__device__ __forceinline__ void mlp_layer(
    const f16* in1, int ld1, int K1,
    const f16* in2, int ld2, int K2,
    const f16* Wlds, const float* bias, int Nout,
    f16* out, int ldOut, int doRelu, int r0, int lane)
{
    const int hf = lane >> 4;
    const int m  = lane & 15;
    const int S  = Nout + 4;
    const int nk1 = K1 / 32, nk2 = K2 / 32;
    v16h af1[8], af2[2];
#pragma unroll
    for (int i = 0; i < nk1; i++) af1[i] = frag_a(in1, ld1, r0 + m, 32*i, hf);
    if (in2) {
#pragma unroll
        for (int i = 0; i < nk2; i++) af2[i] = frag_a(in2, ld2, r0 + m, 32*i, hf);
    }
    const unsigned* W32 = (const unsigned*)Wlds;
    for (int n0 = 0; n0 < Nout; n0 += 16) {
        v8f acc = {0.f,0.f,0.f,0.f,0.f,0.f,0.f,0.f};
#pragma unroll
        for (int i = 0; i < nk1; i++) {
            v16h b = frag_bs(W32, S, 32*i, n0 + m, hf);
            acc = __builtin_amdgcn_wmma_f32_16x16x32_f16(false, af1[i], false, b,
                                                         (short)0, acc, false, false);
        }
        if (in2) {
#pragma unroll
            for (int i = 0; i < nk2; i++) {
                v16h b = frag_bs(W32, S, K1 + 32*i, n0 + m, hf);
                acc = __builtin_amdgcn_wmma_f32_16x16x32_f16(false, af2[i], false, b,
                                                             (short)0, acc, false, false);
            }
        }
        float bv = bias[n0 + m];
#pragma unroll
        for (int rr = 0; rr < 8; rr++) {
            float x = acc[rr] + bv;
            if (doRelu) x = fmaxf(x, 0.0f);
            out[(size_t)(r0 + rr + 8*hf) * ldOut + (n0 + m)] = (f16)x;
        }
    }
}

/* =================== fused MLP over 128-point tiles ===================== */
__global__ __launch_bounds__(256) void mlp_kernel(
    const f16* __restrict__ act0, const f16* __restrict__ denc,
    const f16* __restrict__ Wb,
    const float* __restrict__ b_in,   const float* __restrict__ b_h,
    const float* __restrict__ b_skip, const float* __restrict__ b_sigma,
    const float* __restrict__ b_sem,  const float* __restrict__ b_feat,
    const float* __restrict__ b_rgb1, const float* __restrict__ b_rgb2,
    float* __restrict__ pts)
{
    extern __shared__ f16 smem[];
    f16* X0 = smem;                   /* 128 x 72  */
    f16* Ha = X0 + X0_ELE;            /* 128 x 264 */
    f16* Hb = Ha + H_ELE;             /* 128 x 264 */
    f16* WL = Hb + H_ELE;             /* staged weights, up to 166400 B */

    const int tid  = threadIdx.x;
    const int wave = tid >> 5;
    const int lane = tid & 31;
    const int r0   = wave * 16;
    const int hf   = lane >> 4;
    const int m    = lane & 15;
    const size_t pbase = (size_t)blockIdx.x * TILE_M;

    /* stage posenc features (own rows; padded row stride) */
    {
        const unsigned* src = (const unsigned*)(act0 + (pbase + r0) * 64);
        for (int i = lane; i < 16*32; i += 32) {
            int rr = i >> 5, c = i & 31;
            ((unsigned*)(X0 + (size_t)(r0 + rr) * X0_LD))[c] = src[rr*32 + c];
        }
    }
    __syncthreads();

    stage_weights(Wb + OFF_W_IN, WL, WLDS_OFF_BYTES, 2*SZ_W_IN, tid);
    mlp_layer(X0, X0_LD, 64, (const f16*)0, 0, 0, WL, b_in, 256, Ha, H_LD, 1, r0, lane);
    __syncthreads();
    stage_weights(Wb + OFF_W_H + 0*SZ_W_H1, WL, WLDS_OFF_BYTES, 2*SZ_W_H1, tid);
    mlp_layer(Ha, H_LD, 256, (const f16*)0, 0, 0, WL, b_h + 0, 256, Hb, H_LD, 1, r0, lane);
    __syncthreads();
    stage_weights(Wb + OFF_W_H + 1*SZ_W_H1, WL, WLDS_OFF_BYTES, 2*SZ_W_H1, tid);
    mlp_layer(Hb, H_LD, 256, (const f16*)0, 0, 0, WL, b_h + 256, 256, Ha, H_LD, 1, r0, lane);
    __syncthreads();
    stage_weights(Wb + OFF_W_H + 2*SZ_W_H1, WL, WLDS_OFF_BYTES, 2*SZ_W_H1, tid);
    mlp_layer(Ha, H_LD, 256, (const f16*)0, 0, 0, WL, b_h + 512, 256, Hb, H_LD, 1, r0, lane);
    __syncthreads();
    stage_weights(Wb + OFF_W_SKIP, WL, WLDS_OFF_BYTES, 2*SZ_W_SKIP, tid);
    mlp_layer(Hb, H_LD, 256, X0, X0_LD, 64, WL, b_skip, 256, Ha, H_LD, 1, r0, lane);
    __syncthreads();
    stage_weights(Wb + OFF_W_H + 3*SZ_W_H1, WL, WLDS_OFF_BYTES, 2*SZ_W_H1, tid);
    mlp_layer(Ha, H_LD, 256, (const f16*)0, 0, 0, WL, b_h + 768, 256, Hb, H_LD, 1, r0, lane);
    __syncthreads();
    stage_weights(Wb + OFF_W_H + 4*SZ_W_H1, WL, WLDS_OFF_BYTES, 2*SZ_W_H1, tid);
    mlp_layer(Hb, H_LD, 256, (const f16*)0, 0, 0, WL, b_h + 1024, 256, Ha, H_LD, 1, r0, lane);
    __syncthreads();
    stage_weights(Wb + OFF_W_H + 5*SZ_W_H1, WL, WLDS_OFF_BYTES, 2*SZ_W_H1, tid);
    mlp_layer(Ha, H_LD, 256, (const f16*)0, 0, 0, WL, b_h + 1280, 256, Hb, H_LD, 1, r0, lane);
    __syncthreads();

    /* sigma + sem head (cols: 0=sigma, 1..50=sem), raw */
    stage_weights(Wb + OFF_W_SS, WL, WLDS_OFF_BYTES, 2*SZ_W_SS, tid);
    {
        const unsigned* W32 = (const unsigned*)WL;
        v16h af[8];
#pragma unroll
        for (int i = 0; i < 8; i++) af[i] = frag_a(Hb, H_LD, r0 + m, 32*i, hf);
        for (int n0 = 0; n0 < 64; n0 += 16) {
            v8f acc = {0.f,0.f,0.f,0.f,0.f,0.f,0.f,0.f};
#pragma unroll
            for (int i = 0; i < 8; i++) {
                v16h b = frag_bs(W32, 68, 32*i, n0 + m, hf);
                acc = __builtin_amdgcn_wmma_f32_16x16x32_f16(false, af[i], false, b,
                                                             (short)0, acc, false, false);
            }
            int n = n0 + m;
#pragma unroll
            for (int rr = 0; rr < 8; rr++) {
                size_t p = pbase + r0 + rr + 8*hf;
                float v = acc[rr];
                if (n == 0)       pts[p*56 + 3]           = v + b_sigma[0];
                else if (n <= 50) pts[p*56 + 4 + (n - 1)] = v + b_sem[n - 1];
            }
        }
    }
    __syncthreads();

    /* feat = h @ W_feat + b (no relu) -> Ha ; also stage d_enc into X0 */
    stage_weights(Wb + OFF_W_FEAT, WL, WLDS_OFF_BYTES, 2*SZ_W_FEAT, tid);
    for (int i = lane; i < 16*32; i += 32) {
        int rr = i >> 5, cc = i & 31;
        size_t p = pbase + r0 + rr;
        int ray = (int)(p / SAMP);
        X0[(size_t)(r0 + rr) * X0_LD + cc] = denc[(size_t)ray * 32 + cc];
    }
    mlp_layer(Hb, H_LD, 256, (const f16*)0, 0, 0, WL, b_feat, 256, Ha, H_LD, 0, r0, lane);
    __syncthreads();

    /* hr = relu([feat | d_enc] @ W_rgb1 + b) -> Hb */
    stage_weights(Wb + OFF_W_RGB1, WL, WLDS_OFF_BYTES, 2*SZ_W_RGB1, tid);
    mlp_layer(Ha, H_LD, 256, X0, X0_LD, 32, WL, b_rgb1, 128, Hb, H_LD, 1, r0, lane);
    __syncthreads();

    /* rgb = hr @ W_rgb2 + b (raw; sigmoid at composite) */
    stage_weights(Wb + OFF_W_RGB2, WL, WLDS_OFF_BYTES, 2*SZ_W_RGB2, tid);
    {
        const unsigned* W32 = (const unsigned*)WL;
        v8f acc = {0.f,0.f,0.f,0.f,0.f,0.f,0.f,0.f};
#pragma unroll
        for (int k0 = 0; k0 < 128; k0 += 32) {
            v16h a = frag_a(Hb, H_LD, r0 + m, k0, hf);
            v16h b = frag_bs(W32, 20, k0, m, hf);
            acc = __builtin_amdgcn_wmma_f32_16x16x32_f16(false, a, false, b,
                                                         (short)0, acc, false, false);
        }
#pragma unroll
        for (int rr = 0; rr < 8; rr++) {
            size_t p = pbase + r0 + rr + 8*hf;
            if (m < 3) pts[p*56 + m] = acc[rr] + b_rgb2[m];
        }
    }
}

/* =================== volume-rendering composite ===================== */
__global__ __launch_bounds__(256) void composite_kernel(
    const float* __restrict__ rays, const float* __restrict__ inter,
    const float* __restrict__ zbuf, const float* __restrict__ pts,
    float* __restrict__ out)
{
    int r = blockIdx.x * blockDim.x + threadIdx.x;
    if (r >= N_RAYS) return;
    const float* rb = rays + (size_t)r * 6;
    float dx = rb[3], dy = rb[4], dz = rb[5];
    float scale = sqrtf(dx*dx + dy*dy + dz*dz);

    float nearb[10], farb[10];
#pragma unroll
    for (int b = 0; b < 10; b++) {
        nearb[b] = inter[((size_t)r*10 + b)*4 + 0];
        farb[b]  = inter[((size_t)r*10 + b)*4 + 1];
    }
    float T = 1.0f, rm0 = 0.f, rm1 = 0.f, rm2 = 0.f, depth = 0.f, acc = 0.f;
    float semm[50];
#pragma unroll
    for (int c = 0; c < 50; c++) semm[c] = 0.f;

    for (int s = 0; s < SAMP; s++) {
        float z = zbuf[(size_t)r*SAMP + s];
        float dist = (s < SAMP - 1) ? (zbuf[(size_t)r*SAMP + s + 1] - z)
                                    : 1.0e10f * scale;
        const float* pp = pts + ((size_t)r*SAMP + s) * 56;
        float sigma = pp[3];
        bool anyin = false, bound = false;
#pragma unroll
        for (int b = 0; b < 10; b++) {
            anyin = anyin || (z > nearb[b] && z < farb[b]);
            float d1 = z - farb[b], d2 = nearb[b] - z;
            bound = bound || (d1 > 0.f && d1 < 1e-3f) || (d2 > 0.f && d2 < 1e-3f);
        }
        bool mbbox = (z < DISTC) && !anyin;
        if (mbbox || bound) sigma = 0.f;
        sigma = fmaxf(sigma, 0.f);
        float alpha = 1.0f - expf(-sigma * dist);
        float w = alpha * T;
        T *= (1.0f - alpha + 1e-10f);
        float zt = z / scale;
        rm0 += w / (1.0f + expf(-pp[0]));
        rm1 += w / (1.0f + expf(-pp[1]));
        rm2 += w / (1.0f + expf(-pp[2]));
        depth += w * zt;
        acc   += w;
#pragma unroll
        for (int c = 0; c < 50; c++) semm[c] += w * pp[4 + c];
    }
    float* ob = out + (size_t)r * 55;
    ob[0] = rm0; ob[1] = rm1; ob[2] = rm2; ob[3] = depth; ob[4] = acc;
#pragma unroll
    for (int c = 0; c < 50; c++) ob[5 + c] = semm[c];
}

/* =================== host: workspace layout + launches ===================== */
extern "C" void kernel_launch(void* const* d_in, const int* in_sizes, int n_in,
                              void* d_out, int out_size, void* d_ws, size_t ws_size,
                              hipStream_t stream)
{
    const float* rays    = (const float*)d_in[0];
    const float* inter   = (const float*)d_in[1];
    const float* noise   = (const float*)d_in[2];
    const float* w_in    = (const float*)d_in[3];
    const float* b_in    = (const float*)d_in[4];
    const float* w_h     = (const float*)d_in[5];
    const float* b_h     = (const float*)d_in[6];
    const float* w_skip  = (const float*)d_in[7];
    const float* b_skip  = (const float*)d_in[8];
    const float* w_sigma = (const float*)d_in[9];
    const float* b_sigma = (const float*)d_in[10];
    const float* w_feat  = (const float*)d_in[11];
    const float* b_feat  = (const float*)d_in[12];
    const float* w_rgb1  = (const float*)d_in[13];
    const float* b_rgb1  = (const float*)d_in[14];
    const float* w_rgb2  = (const float*)d_in[15];
    const float* b_rgb2  = (const float*)d_in[16];
    const float* w_sem   = (const float*)d_in[17];
    const float* b_sem   = (const float*)d_in[18];

    char* ws = (char*)d_ws;
    f16*  Wb      = (f16*)ws;
    size_t off    = (size_t)W_TOTAL * sizeof(f16);
    f16*  act0    = (f16*)(ws + off);  off += (size_t)NPTS * 64 * sizeof(f16);
    f16*  dencp   = (f16*)(ws + off);  off += (size_t)N_RAYS * 32 * sizeof(f16);
    float* zbuf   = (float*)(ws + off); off += (size_t)N_RAYS * SAMP * sizeof(float);
    float* pts    = (float*)(ws + off); off += (size_t)NPTS * 56 * sizeof(float);
    (void)ws_size; (void)n_in; (void)in_sizes; (void)out_size;

    hipMemsetAsync(Wb, 0, (size_t)W_TOTAL * sizeof(f16), stream);

    auto grid = [](int n){ return (n + 255) / 256; };
    pack_w<<<grid(63*256), 256, 0, stream>>>(w_in,    Wb + OFF_W_IN,   63, 256, 256, 0);
    for (int i = 0; i < 6; i++)
        pack_w<<<grid(256*256), 256, 0, stream>>>(w_h + (size_t)i*256*256,
                                                  Wb + OFF_W_H + (size_t)i*SZ_W_H1,
                                                  256, 256, 256, 0);
    pack_w<<<grid(319*256), 256, 0, stream>>>(w_skip,  Wb + OFF_W_SKIP, 319, 256, 256, 0);
    pack_w<<<grid(256*1),   256, 0, stream>>>(w_sigma, Wb + OFF_W_SS,   256, 1,   64,  0);
    pack_w<<<grid(256*50),  256, 0, stream>>>(w_sem,   Wb + OFF_W_SS,   256, 50,  64,  1);
    pack_w<<<grid(256*256), 256, 0, stream>>>(w_feat,  Wb + OFF_W_FEAT, 256, 256, 256, 0);
    pack_w<<<grid(283*128), 256, 0, stream>>>(w_rgb1,  Wb + OFF_W_RGB1, 283, 128, 128, 0);
    pack_w<<<grid(128*3),   256, 0, stream>>>(w_rgb2,  Wb + OFF_W_RGB2, 128, 3,   16,  0);

    sample_kernel<<<N_RAYS, 256, 0, stream>>>(rays, inter, noise, act0, dencp, zbuf);

    mlp_kernel<<<NTILES, 256, SMEM_BYTES, stream>>>(act0, dencp, Wb,
                                                    b_in, b_h, b_skip, b_sigma, b_sem,
                                                    b_feat, b_rgb1, b_rgb2, pts);

    composite_kernel<<<grid(N_RAYS), 256, 0, stream>>>(rays, inter, zbuf, pts,
                                                       (float*)d_out);
}
